// TMix_qwen2_37477884625391
// MI455X (gfx1250) — matrix operand order repaired and checked by reading, not compiled, not run on hardware
//
#include <hip/hip_runtime.h>
#include <math.h>

typedef __bf16 v16bf __attribute__((ext_vector_type(16)));
typedef __bf16 v8bf  __attribute__((ext_vector_type(8)));
typedef float  v8f   __attribute__((ext_vector_type(8)));

static constexpr int kB  = 2;
static constexpr int kL  = 2048;
static constexpr int kD  = 2048;
static constexpr int kHD = 128;   // head dim
static constexpr int kQH = 16;    // query heads
static constexpr int kKVH = 4;    // kv heads
static constexpr int kGroups = kQH / kKVH;

// ---------------------------------------------------------------------------
// CDNA5 async copy: global -> LDS, 16 bytes per lane, tracked by ASYNCcnt.
// LDS operand = low 32 bits of the generic pointer (flat->LDS truncation).
// ---------------------------------------------------------------------------
__device__ __forceinline__ void async_copy_b128(void* lds_ptr, const void* gptr) {
  uint32_t lds_addr = (uint32_t)(uintptr_t)lds_ptr;
  uint64_t gaddr    = (uint64_t)(uintptr_t)gptr;
  asm volatile("global_load_async_to_lds_b128 %0, %1, off"
               :: "v"(lds_addr), "v"(gaddr)
               : "memory");
}
// Partial wait: allow up to N async ops (the next tile's copies) to stay in
// flight while guaranteeing older ones completed (in-order completion).
template <int N>
__device__ __forceinline__ void wait_async_le() {
  asm volatile("s_wait_asynccnt %0" :: "i"(N) : "memory");
}
__device__ __forceinline__ void wait_lds_all() {
  asm volatile("s_wait_dscnt 0x0" ::: "memory");
}

// ---------------------------------------------------------------------------
// fp32 -> bf16 conversion (grid-stride)
// ---------------------------------------------------------------------------
__global__ void f32_to_bf16_kernel(const float* __restrict__ in,
                                   __bf16* __restrict__ out, size_t n) {
  size_t i = (size_t)blockIdx.x * blockDim.x + threadIdx.x;
  size_t stride = (size_t)gridDim.x * blockDim.x;
  for (; i < n; i += stride) out[i] = (__bf16)in[i];
}

// ---------------------------------------------------------------------------
// RoPE in place on [B, L, nheads, 128] bf16. angles: [2, L, 128] fp32.
// ---------------------------------------------------------------------------
__global__ void rope_kernel(__bf16* __restrict__ t,
                            const float* __restrict__ angles,
                            int nheads, int Bsz, int L) {
  int total = Bsz * L * nheads * 64;
  int idx = blockIdx.x * blockDim.x + threadIdx.x;
  if (idx >= total) return;
  int i   = idx & 63;
  int h   = (idx >> 6) % nheads;
  int pos = (idx / (64 * nheads)) % L;
  int b   = idx / (64 * nheads * L);
  size_t base = (((size_t)b * L + pos) * nheads + h) * kHD;
  const float* cosp = angles + (size_t)pos * kHD;
  const float* sinp = angles + (size_t)L * kHD + (size_t)pos * kHD;
  float x1 = (float)t[base + i];
  float x2 = (float)t[base + i + 64];
  t[base + i]      = (__bf16)(x1 * cosp[i]      - x2 * sinp[i]);
  t[base + i + 64] = (__bf16)(x2 * cosp[i + 64] + x1 * sinp[i + 64]);
}

// ---------------------------------------------------------------------------
// bf16 GEMM: C[M,N] = A[M,K] @ W[N,K]^T (+ bias[N]).
// Block = 64x64 tile, 128 threads (4 waves), BK = 32.
// Double-buffered LDS; async copies for tile t+1 overlap WMMA on tile t.
// ---------------------------------------------------------------------------
template <typename OutT>
__global__ __launch_bounds__(128)
void gemm_bf16_kernel(const __bf16* __restrict__ A, const __bf16* __restrict__ W,
                      const float* __restrict__ bias, OutT* __restrict__ C,
                      int M, int N, int K) {
  __shared__ __attribute__((aligned(32))) __bf16 sA[2][64 * 32];
  __shared__ __attribute__((aligned(32))) __bf16 sB[2][64 * 32];

  const int tid  = threadIdx.x;
  const int wave = tid >> 5;
  const int lane = tid & 31;
  const int ln   = lane & 15;
  const int lh   = lane >> 4;
  const int m0 = blockIdx.y * 64;
  const int n0 = blockIdx.x * 64;

  v8f acc[4] = {};

  const int r = tid >> 1;         // 0..63
  const int c = (tid & 1) * 16;   // 0 or 16

  auto stage = [&](int buf, int k0) {
    const __bf16* ga = A + (size_t)(m0 + r) * K + k0 + c;
    const __bf16* gb = W + (size_t)(n0 + r) * K + k0 + c;
    async_copy_b128(&sA[buf][r * 32 + c],     ga);
    async_copy_b128(&sA[buf][r * 32 + c + 8], ga + 8);
    async_copy_b128(&sB[buf][r * 32 + c],     gb);
    async_copy_b128(&sB[buf][r * 32 + c + 8], gb + 8);
    if (k0 + 64 < K) {
      __builtin_prefetch(ga + 64, 0, 1);   // warm L2 two tiles ahead
      __builtin_prefetch(gb + 64, 0, 1);
    }
  };

  const int nk = K >> 5;
  stage(0, 0);
  for (int kt = 0; kt < nk; ++kt) {
    const int cur = kt & 1;
    if (kt + 1 < nk) {
      stage(cur ^ 1, (kt + 1) * 32);    // 4 new ops in flight for next tile
      wait_async_le<4>();               // current tile's 4 ops done
    } else {
      wait_async_le<0>();
    }
    __syncthreads();

    const __bf16* tA = sA[cur];
    const __bf16* tB = sB[cur];
    // A fragment (16x32): lane row = ln, elem j -> K = lh*16 + j
    v16bf af;
    {
      const __bf16* ap = &tA[(wave * 16 + ln) * 32];
      v8bf a0 = *reinterpret_cast<const v8bf*>(ap + lh * 16);
      v8bf a1 = *reinterpret_cast<const v8bf*>(ap + lh * 16 + 8);
#pragma unroll
      for (int i = 0; i < 8; ++i) { af[i] = a0[i]; af[8 + i] = a1[i]; }
    }
#pragma unroll
    for (int t = 0; t < 4; ++t) {
      // B fragment (32x16): lane col = ln, elem j -> K = lh*16 + j
      v16bf bfr = *reinterpret_cast<const v16bf*>(&tB[(t * 16 + ln) * 32 + lh * 16]);
      acc[t] = __builtin_amdgcn_wmma_f32_16x16x32_bf16(
          false, af, false, bfr, (short)0, acc[t], false, false);
    }
    __syncthreads();   // all waves done reading 'cur' before it is re-staged
  }

#pragma unroll
  for (int t = 0; t < 4; ++t) {
    const int col = n0 + t * 16 + ln;
    const float bb = bias ? bias[col] : 0.0f;
#pragma unroll
    for (int i = 0; i < 8; ++i) {
      const int row = m0 + wave * 16 + i + 8 * lh;  // C layout: row = vgpr + 8*half
      C[(size_t)row * N + col] = (OutT)(acc[t][i] + bb);
    }
  }
}

// ---------------------------------------------------------------------------
// Flash attention (causal, GQA). Q:[B,L,QH,128] K/V:[B,L,KVH,128] bf16.
// Block = 64 queries of one (b, qh); 4 waves x 16 queries.
// Double-buffered 32-key K/V tiles, async-staged into LDS.
// ---------------------------------------------------------------------------
__global__ __launch_bounds__(128)
void attn_kernel(const __bf16* __restrict__ Qp, const __bf16* __restrict__ Kp,
                 const __bf16* __restrict__ Vp, __bf16* __restrict__ Yb,
                 int Bsz, int L) {
  __shared__ __attribute__((aligned(32))) __bf16 sK[2][32 * kHD];
  __shared__ __attribute__((aligned(32))) __bf16 sV[2][32 * kHD];
  __shared__ __attribute__((aligned(32))) __bf16 sP[4][16 * 32];

  const int tid  = threadIdx.x;
  const int wave = tid >> 5;
  const int lane = tid & 31;
  const int ln   = lane & 15;
  const int lh   = lane >> 4;

  const int qtile = blockIdx.x;
  const int qh    = blockIdx.y;
  const int b     = blockIdx.z;
  const int kvh   = qh / kGroups;
  const int qbase = qtile * 64 + wave * 16;

  const size_t qstride  = (size_t)kQH * kHD;
  const size_t kvstride = (size_t)kKVH * kHD;

  // Load Q fragments for this wave's 16 rows (A layout), kept in registers.
  v16bf aq[4];
  {
    const __bf16* qptr = Qp + ((size_t)b * L + qbase + ln) * qstride + (size_t)qh * kHD;
#pragma unroll
    for (int kc = 0; kc < 4; ++kc) {
      v8bf p0 = *reinterpret_cast<const v8bf*>(qptr + kc * 32 + lh * 16);
      v8bf p1 = *reinterpret_cast<const v8bf*>(qptr + kc * 32 + lh * 16 + 8);
#pragma unroll
      for (int i = 0; i < 8; ++i) { aq[kc][i] = p0[i]; aq[kc][8 + i] = p1[i]; }
    }
  }

  v8f o[8] = {};
  float m_row[8], l_row[8];
#pragma unroll
  for (int i = 0; i < 8; ++i) { m_row[i] = -__builtin_inff(); l_row[i] = 0.0f; }

  const float scale = 0.08838834764831845f;  // 1/sqrt(128)
  const int nkt = (qtile + 1) * 2;           // 32-key tiles needed (causal)

  // Per-thread source slice for K/V staging.
  const int kr = (tid * 32) >> 7;       // key row 0..31
  const int kc = (tid * 32) & 127;      // dim col (multiple of 32)
  auto stage_kv = [&](int buf, int keybase) {
    const size_t g = ((size_t)b * L + keybase + kr) * kvstride + (size_t)kvh * kHD + kc;
#pragma unroll
    for (int u = 0; u < 4; ++u) {
      async_copy_b128(&sK[buf][kr * kHD + kc + u * 8], Kp + g + u * 8);
      async_copy_b128(&sV[buf][kr * kHD + kc + u * 8], Vp + g + u * 8);
    }
    __builtin_prefetch(Kp + g + 32 * kvstride, 0, 1);   // L2 warm two tiles ahead
  };

  stage_kv(0, 0);
  for (int kt = 0; kt < nkt; ++kt) {
    const int cur = kt & 1;
    const int keybase = kt * 32;
    if (kt + 1 < nkt) {
      stage_kv(cur ^ 1, keybase + 32);  // 8 new ops in flight for next tile
      wait_async_le<8>();               // current tile's 8 ops done
    } else {
      wait_async_le<0>();
    }
    __syncthreads();

    const __bf16* tK = sK[cur];
    const __bf16* tV = sV[cur];

    // S = Q @ K^T over this tile: two 16x16 frags (keys 0-15, 16-31).
    v8f s0 = {}, s1 = {};
#pragma unroll
    for (int kcc = 0; kcc < 4; ++kcc) {
      v16bf b0 = *reinterpret_cast<const v16bf*>(&tK[ln * kHD + kcc * 32 + lh * 16]);
      v16bf b1 = *reinterpret_cast<const v16bf*>(&tK[(16 + ln) * kHD + kcc * 32 + lh * 16]);
      s0 = __builtin_amdgcn_wmma_f32_16x16x32_bf16(false, aq[kcc], false, b0, (short)0, s0, false, false);
      s1 = __builtin_amdgcn_wmma_f32_16x16x32_bf16(false, aq[kcc], false, b1, (short)0, s1, false, false);
    }

    // Online softmax, fully unrolled so all vector subscripts are constants.
#pragma unroll
    for (int i = 0; i < 8; ++i) {
      const int q_idx = qbase + i + 8 * lh;
      float v0 = s0[i] * scale;
      float v1 = s1[i] * scale;
      if (keybase + ln > q_idx)       v0 = -__builtin_inff();
      if (keybase + 16 + ln > q_idx)  v1 = -__builtin_inff();
      float mx = fmaxf(v0, v1);
      mx = fmaxf(mx, __shfl_xor(mx, 1, 16));
      mx = fmaxf(mx, __shfl_xor(mx, 2, 16));
      mx = fmaxf(mx, __shfl_xor(mx, 4, 16));
      mx = fmaxf(mx, __shfl_xor(mx, 8, 16));
      const float m_new = fmaxf(m_row[i], mx);
      const float corr  = __expf(m_row[i] - m_new);
      const float p0 = __expf(v0 - m_new);
      const float p1 = __expf(v1 - m_new);
      float rs = p0 + p1;
      rs += __shfl_xor(rs, 1, 16);
      rs += __shfl_xor(rs, 2, 16);
      rs += __shfl_xor(rs, 4, 16);
      rs += __shfl_xor(rs, 8, 16);
      l_row[i] = l_row[i] * corr + rs;
      m_row[i] = m_new;
#pragma unroll
      for (int nc = 0; nc < 8; ++nc) o[nc][i] = o[nc][i] * corr;
      // Stage P (C layout -> row-major LDS) for A-frag reload.
      sP[wave][(i + 8 * lh) * 32 + ln]      = (__bf16)p0;
      sP[wave][(i + 8 * lh) * 32 + 16 + ln] = (__bf16)p1;
    }
    // Wave-internal LDS RAW hazard: wait for our ds_stores before re-reading.
    wait_lds_all();

    // Reload P as A fragment (16 queries x 32 keys).
    v16bf ap;
    {
      const __bf16* pp = &sP[wave][ln * 32];
#pragma unroll
      for (int i = 0; i < 8; ++i) {
        ap[i]     = pp[lh * 16 + i];
        ap[8 + i] = pp[lh * 16 + 8 + i];
      }
    }
    // O += P @ V : 8 dim-chunks of 16.
#pragma unroll
    for (int nc = 0; nc < 8; ++nc) {
      v16bf bfr;
#pragma unroll
      for (int j = 0; j < 16; ++j)
        bfr[j] = tV[(lh * 16 + j) * kHD + nc * 16 + ln];
      o[nc] = __builtin_amdgcn_wmma_f32_16x16x32_bf16(false, ap, false, bfr, (short)0, o[nc], false, false);
    }
    __syncthreads();   // all waves done with 'cur' before it is re-staged
  }

  // Normalize and store Y[B,L,QH,128] bf16.
#pragma unroll
  for (int i = 0; i < 8; ++i) {
    const float inv = 1.0f / l_row[i];
    const int q_idx = qbase + i + 8 * lh;
    __bf16* yp = Yb + ((size_t)b * L + q_idx) * qstride + (size_t)qh * kHD;
#pragma unroll
    for (int nc = 0; nc < 8; ++nc)
      yp[nc * 16 + ln] = (__bf16)(o[nc][i] * inv);
  }
}

// ---------------------------------------------------------------------------
// Host orchestration
// ---------------------------------------------------------------------------
extern "C" void kernel_launch(void* const* d_in, const int* in_sizes, int n_in,
                              void* d_out, int out_size, void* d_ws, size_t ws_size,
                              hipStream_t stream) {
  (void)in_sizes; (void)n_in; (void)out_size; (void)ws_size;
  const float* x      = (const float*)d_in[0];
  const float* angles = (const float*)d_in[1];
  const float* wq     = (const float*)d_in[2];
  const float* bq     = (const float*)d_in[3];
  const float* wk     = (const float*)d_in[4];
  const float* bk     = (const float*)d_in[5];
  const float* wv     = (const float*)d_in[6];
  const float* bv     = (const float*)d_in[7];
  const float* wo     = (const float*)d_in[8];
  float* out = (float*)d_out;

  const int M   = kB * kL;            // 4096
  const int Dkv = kKVH * kHD;         // 512

  char* ws = (char*)d_ws;
  size_t off = 0;
  auto take = [&](size_t bytes) {
    void* p = ws + off;
    off += (bytes + 255) & ~(size_t)255;
    return p;
  };
  __bf16* xb  = (__bf16*)take((size_t)M * kD * 2);
  __bf16* wqb = (__bf16*)take((size_t)kD * kD * 2);
  __bf16* wkb = (__bf16*)take((size_t)Dkv * kD * 2);
  __bf16* wvb = (__bf16*)take((size_t)Dkv * kD * 2);
  __bf16* wob = (__bf16*)take((size_t)kD * kD * 2);
  __bf16* Qp  = (__bf16*)take((size_t)M * kD * 2);
  __bf16* Kp  = (__bf16*)take((size_t)M * Dkv * 2);
  __bf16* Vp  = (__bf16*)take((size_t)M * Dkv * 2);
  __bf16* Yb  = (__bf16*)take((size_t)M * kD * 2);

  // 1) convert activations + weights to bf16
  f32_to_bf16_kernel<<<2048, 256, 0, stream>>>(x,  xb,  (size_t)M * kD);
  f32_to_bf16_kernel<<<2048, 256, 0, stream>>>(wq, wqb, (size_t)kD * kD);
  f32_to_bf16_kernel<<<512,  256, 0, stream>>>(wk, wkb, (size_t)Dkv * kD);
  f32_to_bf16_kernel<<<512,  256, 0, stream>>>(wv, wvb, (size_t)Dkv * kD);
  f32_to_bf16_kernel<<<2048, 256, 0, stream>>>(wo, wob, (size_t)kD * kD);

  dim3 blk(128);
  // 2) projections (y = x @ W^T + b)
  gemm_bf16_kernel<__bf16><<<dim3(kD / 64,  M / 64), blk, 0, stream>>>(xb, wqb, bq, Qp, M, kD,  kD);
  gemm_bf16_kernel<__bf16><<<dim3(Dkv / 64, M / 64), blk, 0, stream>>>(xb, wkb, bk, Kp, M, Dkv, kD);
  gemm_bf16_kernel<__bf16><<<dim3(Dkv / 64, M / 64), blk, 0, stream>>>(xb, wvb, bv, Vp, M, Dkv, kD);

  // 3) RoPE on Q and K
  rope_kernel<<<(kB * kL * kQH * 64 + 255) / 256, 256, 0, stream>>>(Qp, angles, kQH, kB, kL);
  rope_kernel<<<(kB * kL * kKVH * 64 + 255) / 256, 256, 0, stream>>>(Kp, angles, kKVH, kB, kL);

  // 4) causal flash attention with GQA
  attn_kernel<<<dim3(kL / 64, kQH, kB), blk, 0, stream>>>(Qp, Kp, Vp, Yb, kB, kL);

  // 5) output projection -> fp32 d_out
  gemm_bf16_kernel<float><<<dim3(kD / 64, M / 64), blk, 0, stream>>>(Yb, wob, nullptr, out, M, kD, kD);
}
